// SpatialLiDAREncoder_23905787969766
// MI455X (gfx1250) — compile-verified, matrix operand order checked by reading, not run on hardware
//
#include <hip/hip_runtime.h>
#include <hip/hip_bf16.h>

typedef __attribute__((ext_vector_type(16))) _Float16 v16h;
typedef __attribute__((ext_vector_type(8)))  float    v8f;

#define B_SZ   4
#define N_PTS  120000
#define C_IN   4
#define FEAT   128
#define HH     128
#define WW     128
#define EPSF   1e-5f
#define NT     ((B_SZ * N_PTS) / 16)   // 30000 tiles of 16 points
#define WAVES_PER_BLOCK 4

__device__ __forceinline__ v8f wmma_f16(v16h a, v16h b, v8f c) {
    return __builtin_amdgcn_wmma_f32_16x16x32_f16(
        /*neg_a=*/false, a, /*neg_b=*/false, b,
        /*c_mod=*/(short)0, c, /*reuse_a=*/false, /*reuse_b=*/false);
}

// Load a 16-half fragment stored contiguously (pre-swizzled weights).
__device__ __forceinline__ v16h ld_frag(const _Float16* p) {
    v16h r;
    *(uint4*)&r       = *(const uint4*)p;
    *((uint4*)&r + 1) = *(const uint4*)(p + 8);
    return r;
}

// Build an A-fragment for this lane from a row-major activation row.
// Layout (16-bit A 16x32): lanes 0-15 hold K = {0..7, 16..23}, lanes 16-31 hold K = {8..15, 24..31}.
__device__ __forceinline__ v16h ld_act_frag(const _Float16* row, int kstep, int hiHalf) {
    int base = kstep * 32 + (hiHalf ? 8 : 0);
    v16h r;
    *(uint4*)&r       = *(const uint4*)(row + base);
    *((uint4*)&r + 1) = *(const uint4*)(row + base + 16);
    return r;
}

// Map a fragment element (lane, h) to its K within a 32-wide K-step
// (mirror of A layout; N = lane & 15).
__device__ __forceinline__ int frag_klocal(int lane, int h) {
    return (h < 8) ? ((lane < 16 ? 0 : 8) + h)
                   : ((lane < 16 ? 16 : 24) + (h - 8));
}

__global__ void __launch_bounds__(WAVES_PER_BLOCK * 32)
pillar_fused_kernel(const float* __restrict__ points,
                    const float* __restrict__ W1, const float* __restrict__ b1,
                    const float* __restrict__ g1, const float* __restrict__ be1,
                    const float* __restrict__ m1, const float* __restrict__ v1,
                    const float* __restrict__ W2, const float* __restrict__ b2,
                    const float* __restrict__ g2, const float* __restrict__ be2,
                    const float* __restrict__ m2, const float* __restrict__ v2,
                    const float* __restrict__ W3, const float* __restrict__ b3,
                    const float* __restrict__ g3, const float* __restrict__ be3,
                    const float* __restrict__ m3, const float* __restrict__ v3,
                    float* __restrict__ bev)
{
    // ---- LDS: pre-swizzled f16 weight fragments + folded BN tables + per-wave scratch ----
    __shared__ __align__(16) _Float16 sW1[4][512];          //  4 KB  (ntile)
    __shared__ __align__(16) _Float16 sW2[8][2][512];       // 16 KB  (ntile, kstep)
    __shared__ __align__(16) _Float16 sW3[8][4][512];       // 32 KB
    __shared__ float sS1[64],  sT1[64];
    __shared__ float sS2[128], sT2[128];
    __shared__ float sS3[128], sT3[128];
    __shared__ __align__(16) _Float16 sAct1[WAVES_PER_BLOCK][16][64];   //  8 KB
    __shared__ __align__(16) _Float16 sAct2[WAVES_PER_BLOCK][16][128];  // 16 KB
    __shared__ __align__(16) int      sOff[WAVES_PER_BLOCK][16];

    const int tid = threadIdx.x;
    const int nthreads = blockDim.x;

    // ---- one-time setup: fold BN into scale/bias, swizzle weights to fragment order ----
    for (int c = tid; c < 64; c += nthreads) {
        float s = g1[c] * rsqrtf(v1[c] + EPSF);
        sS1[c] = s; sT1[c] = (b1[c] - m1[c]) * s + be1[c];
    }
    for (int c = tid; c < 128; c += nthreads) {
        float s2 = g2[c] * rsqrtf(v2[c] + EPSF);
        sS2[c] = s2; sT2[c] = (b2[c] - m2[c]) * s2 + be2[c];
        float s3 = g3[c] * rsqrtf(v3[c] + EPSF);
        sS3[c] = s3; sT3[c] = (b3[c] - m3[c]) * s3 + be3[c];
    }
    for (int i = tid; i < 4 * 512; i += nthreads) {          // W1: (64,4), K padded to 32
        int t = i >> 9, e = i & 511, lane = e >> 4, h = e & 15;
        int n = lane & 15, kl = frag_klocal(lane, h);
        sW1[t][e] = (_Float16)((kl < C_IN) ? W1[(t * 16 + n) * C_IN + kl] : 0.0f);
    }
    for (int i = tid; i < 16 * 512; i += nthreads) {         // W2: (128,64)
        int f = i >> 9, e = i & 511, t = f >> 1, ks = f & 1;
        int lane = e >> 4, h = e & 15;
        int n = lane & 15, k = ks * 32 + frag_klocal(lane, h);
        sW2[t][ks][e] = (_Float16)W2[(t * 16 + n) * 64 + k];
    }
    for (int i = tid; i < 32 * 512; i += nthreads) {         // W3: (128,128)
        int f = i >> 9, e = i & 511, t = f >> 2, ks = f & 3;
        int lane = e >> 4, h = e & 15;
        int n = lane & 15, k = ks * 32 + frag_klocal(lane, h);
        sW3[t][ks][e] = (_Float16)W3[(t * 16 + n) * 128 + k];
    }
    __syncthreads();

    const int wave   = tid >> 5;
    const int lane   = tid & 31;
    const int m16    = lane & 15;        // point row for A-frags / BN column
    const int hiHalf = lane >> 4;        // 0: lanes 0-15, 1: lanes 16-31
    const int mbase  = hiHalf ? 8 : 0;   // D-frag row base for this lane half

    const int gwave  = blockIdx.x * WAVES_PER_BLOCK + wave;
    const int nwaves = gridDim.x * WAVES_PER_BLOCK;

    for (int tile = gwave; tile < NT; tile += nwaves) {
        const int g0 = tile * 16;                         // N % 16 == 0 -> tile never straddles batches
        const int b  = g0 / N_PTS;

        // ---- load 16 points (one per lane mod 16), build layer-1 A fragment ----
        float4 p = *(const float4*)(points + (size_t)(g0 + m16) * C_IN);
        v16h a1 = {};
        if (!hiHalf) {   // lanes 0-15 carry K=0..3; all other K slots stay zero
            a1[0] = (_Float16)p.x; a1[1] = (_Float16)p.y;
            a1[2] = (_Float16)p.z; a1[3] = (_Float16)p.w;
        }

        // ---- per-point BEV segment offset (lanes 0-15 write) ----
        if (!hiHalf) {
            float xn = (p.x + 50.0f) * 0.01f;
            float yn = (p.y + 50.0f) * 0.01f;
            bool valid = (xn >= 0.0f) & (xn <= 1.0f) & (yn >= 0.0f) & (yn <= 1.0f);
            int gx = min(max((int)(xn * (float)(WW - 1)), 0), WW - 1);
            int gy = min(max((int)(yn * (float)(HH - 1)), 0), HH - 1);
            sOff[wave][m16] = valid ? (b * FEAT * HH * WW + gy * WW + gx) : -1;
        }

        // ---- layer 1: 4 -> 64  (4 WMMAs) ----
        #pragma unroll
        for (int t = 0; t < 4; ++t) {
            v8f acc = {};
            acc = wmma_f16(a1, ld_frag(&sW1[t][lane * 16]), acc);
            int c = t * 16 + m16;
            float s = sS1[c], tb = sT1[c];
            #pragma unroll
            for (int j = 0; j < 8; ++j)
                sAct1[wave][mbase + j][c] = (_Float16)fmaxf(acc[j] * s + tb, 0.0f);
        }

        // ---- layer 2: 64 -> 128  (16 WMMAs) ----
        const _Float16* row1 = &sAct1[wave][m16][0];
        v16h a2_0 = ld_act_frag(row1, 0, hiHalf);
        v16h a2_1 = ld_act_frag(row1, 1, hiHalf);
        #pragma unroll
        for (int t = 0; t < 8; ++t) {
            v8f acc = {};
            acc = wmma_f16(a2_0, ld_frag(&sW2[t][0][lane * 16]), acc);
            acc = wmma_f16(a2_1, ld_frag(&sW2[t][1][lane * 16]), acc);
            int c = t * 16 + m16;
            float s = sS2[c], tb = sT2[c];
            #pragma unroll
            for (int j = 0; j < 8; ++j)
                sAct2[wave][mbase + j][c] = (_Float16)fmaxf(acc[j] * s + tb, 0.0f);
        }

        // ---- layer 3: 128 -> 128  (32 WMMAs) + fused scatter-max ----
        const _Float16* row2 = &sAct2[wave][m16][0];
        v16h a3_0 = ld_act_frag(row2, 0, hiHalf);
        v16h a3_1 = ld_act_frag(row2, 1, hiHalf);
        v16h a3_2 = ld_act_frag(row2, 2, hiHalf);
        v16h a3_3 = ld_act_frag(row2, 3, hiHalf);

        int offs[8];
        {
            const int* ob = &sOff[wave][mbase];   // this lane half's 8 point rows
            #pragma unroll
            for (int j = 0; j < 8; ++j) offs[j] = ob[j];
        }

        #pragma unroll
        for (int t = 0; t < 8; ++t) {
            v8f acc = {};
            acc = wmma_f16(a3_0, ld_frag(&sW3[t][0][lane * 16]), acc);
            acc = wmma_f16(a3_1, ld_frag(&sW3[t][1][lane * 16]), acc);
            acc = wmma_f16(a3_2, ld_frag(&sW3[t][2][lane * 16]), acc);
            acc = wmma_f16(a3_3, ld_frag(&sW3[t][3][lane * 16]), acc);
            int c = t * 16 + m16;
            float s = sS3[c], tb = sT3[c];
            unsigned int* outc = (unsigned int*)bev + (size_t)c * (HH * WW);
            #pragma unroll
            for (int j = 0; j < 8; ++j) {
                float y = fmaxf(acc[j] * s + tb, 0.0f);   // y >= 0 -> uint-max == float-max
                if (offs[j] >= 0)
                    atomicMax(outc + offs[j], __float_as_uint(y));
            }
        }
    }
}

__global__ void bev_zero_kernel(float4* __restrict__ out, int n4) {
    int i = blockIdx.x * blockDim.x + threadIdx.x;
    if (i < n4) out[i] = make_float4(0.f, 0.f, 0.f, 0.f);
}

extern "C" void kernel_launch(void* const* d_in, const int* in_sizes, int n_in,
                              void* d_out, int out_size, void* d_ws, size_t ws_size,
                              hipStream_t stream) {
    const float* points = (const float*)d_in[0];
    const float* W1  = (const float*)d_in[1];
    const float* b1  = (const float*)d_in[2];
    const float* g1  = (const float*)d_in[3];
    const float* be1 = (const float*)d_in[4];
    const float* m1  = (const float*)d_in[5];
    const float* v1  = (const float*)d_in[6];
    const float* W2  = (const float*)d_in[7];
    const float* b2  = (const float*)d_in[8];
    const float* g2  = (const float*)d_in[9];
    const float* be2 = (const float*)d_in[10];
    const float* m2  = (const float*)d_in[11];
    const float* v2  = (const float*)d_in[12];
    const float* W3  = (const float*)d_in[13];
    const float* b3  = (const float*)d_in[14];
    const float* g3  = (const float*)d_in[15];
    const float* be3 = (const float*)d_in[16];
    const float* m3  = (const float*)d_in[17];
    const float* v3  = (const float*)d_in[18];
    float* bev = (float*)d_out;

    // Zero-init BEV grid (empty cells must be 0; enables unsigned atomic-max scatter).
    int n4 = out_size / 4;                       // 8,388,608 floats -> 2,097,152 float4
    bev_zero_kernel<<<(n4 + 255) / 256, 256, 0, stream>>>((float4*)bev, n4);

    // Fused MLP (WMMA f16) + BN/ReLU + scatter-max.
    pillar_fused_kernel<<<256, WAVES_PER_BLOCK * 32, 0, stream>>>(
        points, W1, b1, g1, be1, m1, v1,
        W2, b2, g2, be2, m2, v2,
        W3, b3, g3, be3, m3, v3, bev);
}
